// Attention_34806414967022
// MI455X (gfx1250) — matrix operand order repaired and checked by reading, not compiled, not run on hardware
//
#include <hip/hip_runtime.h>
#include <hip/hip_bf16.h>
#include <math.h>
#include <stdint.h>

// ---------------- problem constants ----------------
constexpr int B_  = 8;
constexpr int C_  = 128;
constexpr int C3  = 384;
constexpr int HW  = 128;
constexpr int L_  = HW * HW;     // 16384
constexpr int KP  = 136;         // bf16 K-stride (128 + 8) -> 272B rows, 16B aligned, bank-rotated
constexpr int KP2 = 72;          // bf16 K-stride for 64-wide chunks (144B rows)
constexpr int NP  = 68;          // f32 N-stride for 64-col tiles
constexpr int WAVES = 8;         // 256 threads / wave32
constexpr int KSLICES = 16;      // K-split for the attention-logits GEMM

// ---------------- workspace layout (float slots) ----------------
constexpr size_t W_QKV0H = 0;                                   // 8*384*16384 bf16 (pre-conv)
constexpr size_t W_QKVH  = 25165824;                            // 8*384*16384 bf16 (post-conv)
constexpr size_t W_CONVP = 50331648;                            // 49152 f32 block partials
constexpr size_t W_IDX   = W_CONVP + 49152;                     // 128 ints
constexpr size_t W_INVQ  = W_IDX + 128;                         // 8*128
constexpr size_t W_INVK  = W_INVQ + 1024;
constexpr size_t W_SUMQ  = W_INVK + 1024;
constexpr size_t W_SUMK  = W_SUMQ + 1024;
constexpr size_t W_ATTN  = W_SUMK + 1024;                       // 8*4*48*48
constexpr size_t W_ATTNP = W_ATTN + 73728;                      // 16 * 8*4*48*48 partials
constexpr size_t W_WH    = W_ATTNP + (size_t)KSLICES * 73728;   // 49152 bf16 weights

// ---------------- types ----------------
typedef __attribute__((ext_vector_type(16))) __bf16 v16bf;
typedef __attribute__((ext_vector_type(8)))  __bf16 v8bf;
typedef __attribute__((ext_vector_type(8)))  float  v8f;
typedef __attribute__((ext_vector_type(4)))  unsigned u32x4;
typedef __attribute__((ext_vector_type(8)))  unsigned u32x8;

union FragAB { v16bf v; __uint128_t u[2]; };
union Pack4  { __bf16 h[4]; unsigned long long u; };

// A operand: row-major [16][kp] bf16 tile. Lane ln: row = ln&15, K = 8*(ln>>4) + {0..7, 16..23}
__device__ __forceinline__ v16bf load_frag_a(const __bf16* tile, int kp) {
  const int ln = threadIdx.x & 31;
  const __bf16* p = tile + (ln & 15) * kp + ((ln >> 4) << 3);
  FragAB f;
  f.u[0] = *(const __uint128_t*)(p);
  f.u[1] = *(const __uint128_t*)(p + 16);
  return f.v;
}
// B operand from N-major [16][kp] tile. Lane ln: n = ln&15, K = 8*(ln>>4) + {0..7, 16..23}
__device__ __forceinline__ v16bf load_frag_b(const __bf16* tileT, int kp) {
  const int ln = threadIdx.x & 31;
  const __bf16* p = tileT + (ln & 15) * kp + ((ln >> 4) << 3);
  FragAB f;
  f.u[0] = *(const __uint128_t*)(p);
  f.u[1] = *(const __uint128_t*)(p + 16);
  return f.v;
}
__device__ __forceinline__ void frag_coords(int& col, int& rb) {
  const int ln = threadIdx.x & 31;
  col = ln & 15;
  rb  = (ln >> 4) * 8;
}
__device__ __forceinline__ v8f wmma_bf16(v16bf a, v16bf b, v8f c) {
  return __builtin_amdgcn_wmma_f32_16x16x32_bf16(false, a, false, b, (short)0, c, false, false);
}

// ---------------- CDNA5 async copy: global -> LDS, 16B per lane ----------------
__device__ __forceinline__ void async_copy_b128(unsigned lds_byte_off, const void* gaddr) {
  unsigned long long ga = (unsigned long long)(uintptr_t)gaddr;
  asm volatile("global_load_async_to_lds_b128 %0, %1, off"
               :: "v"(lds_byte_off), "v"(ga) : "memory");
}
__device__ __forceinline__ void async_wait0() {
  asm volatile("s_wait_asynccnt 0x0" ::: "memory");
}

// ---------------- CDNA5 Tensor Data Mover: 2D bf16 tile -> padded LDS rows ----------------
__device__ __forceinline__ void tdm_load_2d(const __bf16* gsrc, unsigned lds_byte_off,
                                            int rows, int cols, int gstride,
                                            unsigned pad_interval_code, unsigned pad_amount_code) {
  unsigned long long ga = (unsigned long long)(uintptr_t)gsrc;
  u32x4 g0;
  g0[0] = 1u;
  g0[1] = lds_byte_off;
  g0[2] = (unsigned)(ga & 0xffffffffull);
  g0[3] = (unsigned)((ga >> 32) & 0x01ffffffull) | (2u << 30);
  u32x8 g1;
  g1[0] = (1u << 16) | (1u << 20)
        | (pad_interval_code << 22) | (pad_amount_code << 25);
  g1[1] = ((unsigned)cols & 0xffffu) << 16;
  g1[2] = (((unsigned)cols >> 16) & 0xffffu) | (((unsigned)rows & 0xffffu) << 16);
  g1[3] = (((unsigned)rows >> 16) & 0xffffu) | (((unsigned)cols & 0xffffu) << 16);
  g1[4] = (unsigned)rows & 0xffffu;
  g1[5] = (unsigned)gstride;
  g1[6] = 0u;
  g1[7] = 0u;
  asm volatile("tensor_load_to_lds %0, %1" :: "s"(g0), "s"(g1) : "memory");
}

// Generic LDS-resident tile GEMM: D(MT*16 x NT*16) = A(MxK) * B(KxN), K = KT*32.
template<int MT, int NT, int KT, int KPA>
__device__ __forceinline__ void gemm_tiles(const __bf16* A, const __bf16* Bt, v8f* acc) {
  const int w = threadIdx.x >> 5;
  constexpr int TOT = MT * NT;
  constexpr int PER = TOT / WAVES;
#pragma unroll
  for (int q = 0; q < PER; ++q) {
    const int s  = w * PER + q;
    const int mt = s / NT, nt = s % NT;
    v8f c;
#pragma unroll
    for (int r = 0; r < 8; ++r) c[r] = 0.0f;
#pragma unroll
    for (int kt = 0; kt < KT; ++kt) {
      v16bf av = load_frag_a(A  + mt * 16 * KPA + kt * 32, KPA);
      v16bf bv = load_frag_b(Bt + nt * 16 * KPA + kt * 32, KPA);
      c = wmma_bf16(av, bv, c);
    }
    acc[q] = c;
  }
}

// ================= K0: weights -> bf16 mirror =================
__global__ void k_cvtw(const float* __restrict__ wg, const float* __restrict__ wd,
                       const float* __restrict__ wu, const float* __restrict__ wp,
                       __bf16* __restrict__ wh) {
  int t = blockIdx.x * 256 + threadIdx.x;
  if      (t < 16384) wh[t] = (__bf16)wg[t];
  else if (t < 24576) wh[t] = (__bf16)wd[t - 16384];
  else if (t < 32768) wh[t] = (__bf16)wu[t - 24576];
  else if (t < 49152) wh[t] = (__bf16)wp[t - 32768];
}

// ================= K1: qkv = W_qkv(384x128) @ x(b,128,L) -> bf16 =================
__global__ void k_qkv(const float* __restrict__ x, const float* __restrict__ w_qkv,
                      __bf16* __restrict__ qkv0h) {
  __shared__ alignas(16) __bf16 As[64 * KP];
  __shared__ alignas(16) __bf16 Bs[64 * KP];
  const int l0 = blockIdx.x * 64;
  const int m0 = blockIdx.y * 64;
  const int b  = blockIdx.z;
  const int tid = threadIdx.x;
  // stage A (w_qkv rows) with float4 reads + packed 8B LDS writes
  for (int e = tid; e < 2048; e += 256) {
    int r = e >> 5, cq = e & 31;
    float4 wv = *(const float4*)(w_qkv + (m0 + r) * C_ + cq * 4);
    Pack4 pk;
    pk.h[0] = (__bf16)wv.x; pk.h[1] = (__bf16)wv.y;
    pk.h[2] = (__bf16)wv.z; pk.h[3] = (__bf16)wv.w;
    *(unsigned long long*)(As + r * KP + cq * 4) = pk.u;
  }
  // stage B (x tile, transposed) with float4 reads
  for (int e = tid; e < 128 * 16; e += 256) {
    int k = e >> 4, nq = e & 15;
    float4 xv = *(const float4*)(x + ((size_t)(b * C_ + k)) * L_ + l0 + nq * 4);
    int n = nq * 4;
    Bs[(n + 0) * KP + k] = (__bf16)xv.x;
    Bs[(n + 1) * KP + k] = (__bf16)xv.y;
    Bs[(n + 2) * KP + k] = (__bf16)xv.z;
    Bs[(n + 3) * KP + k] = (__bf16)xv.w;
  }
  __syncthreads();
  v8f acc[2];
  gemm_tiles<4, 4, 4, KP>(As, Bs, acc);
  __syncthreads();                      // A/B reads done -> reuse As as store-stage
  int col, rb; frag_coords(col, rb);
  const int w = tid >> 5;
  __bf16* sOut = As;                    // 64 x 64 bf16
#pragma unroll
  for (int q = 0; q < 2; ++q) {
    int s = w * 2 + q, mt = s >> 2, nt = s & 3;
#pragma unroll
    for (int r = 0; r < 8; ++r)
      sOut[(mt * 16 + rb + r) * 64 + nt * 16 + col] = (__bf16)acc[q][r];
  }
  __syncthreads();
  // vectorized b128 stores: 64 rows x 8 segments of 8 bf16
  for (int seg = tid; seg < 512; seg += 256) {
    int m = seg >> 3, sg = seg & 7;
    *(__uint128_t*)(qkv0h + ((size_t)(b * C3 + m0 + m)) * L_ + l0 + sg * 8) =
        *(const __uint128_t*)(sOut + m * 64 + sg * 8);
  }
}

// ================= K2: depthwise 3x3 SAME conv, 4 pixels/thread, 8B stores =================
// Each block covers 1024 consecutive positions of ONE (b, channel) row; exact f32 partial
// sums per block feed the argsort (deterministic, no rounding before the rank decision).
__global__ void k_dwconv(const __bf16* __restrict__ qkv0h, const float* __restrict__ w_dw,
                         __bf16* __restrict__ qkvh, float* __restrict__ convP) {
  const size_t id4 = (size_t)blockIdx.x * 256 + threadIdx.x;   // quad index
  const int p4 = (int)(id4 & (L_ / 4 - 1));
  const size_t bc = id4 >> 12;                                 // / (L_/4)
  const int o = (int)(bc % C3);
  const int y  = p4 >> 5;
  const int x0 = (p4 & 31) * 4;
  const __bf16* in = qkv0h + bc * L_;
  const float* wp = w_dw + o * 9;
  float s0 = 0, s1 = 0, s2 = 0, s3 = 0;
#pragma unroll
  for (int ky = 0; ky < 3; ++ky) {
    int iy = y + ky - 1;
    if (iy < 0 || iy > 127) continue;
    const __bf16* rowp = in + iy * 128;
    float v[6];
#pragma unroll
    for (int i = 0; i < 6; ++i) {
      int ix = x0 - 1 + i;
      v[i] = (ix < 0 || ix > 127) ? 0.0f : (float)rowp[ix];
    }
    float w0 = wp[ky * 3 + 0], w1 = wp[ky * 3 + 1], w2 = wp[ky * 3 + 2];
    s0 += v[0] * w0 + v[1] * w1 + v[2] * w2;
    s1 += v[1] * w0 + v[2] * w1 + v[3] * w2;
    s2 += v[2] * w0 + v[3] * w1 + v[4] * w2;
    s3 += v[3] * w0 + v[4] * w1 + v[5] * w2;
  }
  Pack4 pk;
  pk.h[0] = (__bf16)s0; pk.h[1] = (__bf16)s1;
  pk.h[2] = (__bf16)s2; pk.h[3] = (__bf16)s3;
  *(unsigned long long*)(qkvh + bc * L_ + p4 * 4) = pk.u;
  __shared__ float red[256];
  red[threadIdx.x] = s0 + s1 + s2 + s3; __syncthreads();
  for (int off = 128; off > 0; off >>= 1) {
    if ((int)threadIdx.x < off) red[threadIdx.x] += red[threadIdx.x + off];
    __syncthreads();
  }
  if (threadIdx.x == 0) convP[blockIdx.x] = red[0];
}

// ================= K4: exact channel means (f32 partials) + stable rank =================
__global__ void k_rank(const float* __restrict__ convP, int* __restrict__ idxp) {
  __shared__ float m[128];
  const int t = threadIdx.x;
  float s = 0.0f;
  for (int b = 0; b < B_; ++b)
    for (int p = 0; p < 16; ++p)
      s += convP[((size_t)b * C3 + t) * 16 + p];
  m[t] = s;
  __syncthreads();
  float mi = m[t]; int r = 0;
  for (int j = 0; j < 128; ++j)
    r += (m[j] > mi) || (m[j] == mi && j < t);
  idxp[r] = t;
}

// ================= K5: inverse L2 norms + row sums (16B vector loads) =================
__global__ void k_norms(const __bf16* __restrict__ qkvh, float* __restrict__ invq,
                        float* __restrict__ invk, float* __restrict__ sumq,
                        float* __restrict__ sumk) {
  __shared__ float r1[256], r2[256];
  const int ch = blockIdx.x, sel = blockIdx.y, b = blockIdx.z;
  const v8bf* row8 = (const v8bf*)(qkvh + ((size_t)(b * C3 + sel * C_ + ch)) * L_);
  float ss = 0.0f, sm = 0.0f;
  for (int e = threadIdx.x; e < L_ / 8; e += 256) {
    v8bf v = row8[e];
#pragma unroll
    for (int i = 0; i < 8; ++i) { float f = (float)v[i]; ss += f * f; sm += f; }
  }
  r1[threadIdx.x] = ss; r2[threadIdx.x] = sm; __syncthreads();
  for (int off = 128; off > 0; off >>= 1) {
    if ((int)threadIdx.x < off) {
      r1[threadIdx.x] += r1[threadIdx.x + off];
      r2[threadIdx.x] += r2[threadIdx.x + off];
    }
    __syncthreads();
  }
  if (threadIdx.x == 0) {
    float inv = 1.0f / fmaxf(sqrtf(r1[0]), 1e-12f);
    if (sel == 0) { invq[b * C_ + ch] = inv; sumq[b * C_ + ch] = r2[0]; }
    else          { invk[b * C_ + ch] = inv; sumk[b * C_ + ch] = r2[0]; }
  }
}

// ================= K6: partial S = q . k^T over one K-slice (async-to-LDS staged) =================
template<int GT>
__device__ void attn_group(int b, int start, int kbase, __bf16* qB, __bf16* kB,
                           const __bf16* __restrict__ qkvh, const int* __restrict__ idxp,
                           float* __restrict__ P) {
  constexpr int TOT = GT * GT;
  constexpr int G   = GT * 16;
  const int w = threadIdx.x >> 5;
  const unsigned qB_off = (unsigned)(uintptr_t)qB;
  const unsigned kB_off = (unsigned)(uintptr_t)kB;
  v8f acc[2];
#pragma unroll
  for (int q = 0; q < 2; ++q)
#pragma unroll
    for (int r = 0; r < 8; ++r) acc[q][r] = 0.0f;

  for (int kk = kbase; kk < kbase + L_ / KSLICES; kk += 64) {
    for (int sgi = threadIdx.x; sgi < G * 16; sgi += 256) {
      int mat = sgi / (G * 8);
      int rem = sgi - mat * (G * 8);
      int r  = rem >> 3;
      int sg = rem & 7;
      int ch = idxp[start + r];
      const __bf16* src = qkvh + ((size_t)(b * C3 + mat * 128 + ch)) * L_ + kk + sg * 8;
      unsigned dst = (mat ? kB_off : qB_off) + (unsigned)(r * KP2 * 2 + sg * 16);
      async_copy_b128(dst, src);
    }
    async_wait0();
    __syncthreads();
    {
      int q = 0;
      for (int s = w; s < TOT; s += WAVES, ++q) {
        int mt = s / GT, nt = s % GT;
#pragma unroll
        for (int kt = 0; kt < 2; ++kt) {
          v16bf av = load_frag_a(qB + mt * 16 * KP2 + kt * 32, KP2);
          v16bf bv = load_frag_b(kB + nt * 16 * KP2 + kt * 32, KP2);
          acc[q] = wmma_bf16(av, bv, acc[q]);
        }
      }
    }
    __syncthreads();
  }
  int col, rb; frag_coords(col, rb);
  {
    int q = 0;
    for (int s = w; s < TOT; s += WAVES, ++q) {
      int mt = s / GT, nt = s % GT;
#pragma unroll
      for (int r = 0; r < 8; ++r)
        P[(mt * 16 + rb + r) * 48 + nt * 16 + col] = acc[q][r];
    }
  }
}

__global__ void k_attn(const __bf16* __restrict__ qkvh, const int* __restrict__ idxp,
                       float* __restrict__ attnP) {
  __shared__ alignas(16) __bf16 qB[48 * KP2];
  __shared__ alignas(16) __bf16 kB[48 * KP2];
  const int gi = blockIdx.x, ks = blockIdx.y, b = blockIdx.z;
  const int sts[4] = {0, 16, 48, 80};
  const int kbase = ks * (L_ / KSLICES);
  float* P = attnP + (size_t)((b * KSLICES + ks) * 4 + gi) * 2304;
  switch (gi) {
    case 0:  attn_group<1>(b, sts[0], kbase, qB, kB, qkvh, idxp, P); break;
    case 1:  attn_group<2>(b, sts[1], kbase, qB, kB, qkvh, idxp, P); break;
    case 2:  attn_group<2>(b, sts[2], kbase, qB, kB, qkvh, idxp, P); break;
    default: attn_group<3>(b, sts[3], kbase, qB, kB, qkvh, idxp, P); break;
  }
}

// ================= K6b: deterministic slice-reduction + folded normalization =================
__global__ void k_attnred(const float* __restrict__ attnP, const int* __restrict__ idxp,
                          const float* __restrict__ invq, const float* __restrict__ invk,
                          float* __restrict__ attnS) {
  const int e = blockIdx.x * 256 + threadIdx.x;
  if (e >= B_ * 4 * 2304) return;
  const int bg = e / 2304, rem = e - bg * 2304;
  const int b = bg >> 2, gi = bg & 3;
  const int m = rem / 48, n = rem - m * 48;
  const int sts[4] = {0, 16, 48, 80};
  float s = 0.0f;
  for (int ks = 0; ks < KSLICES; ++ks)
    s += attnP[(size_t)((b * KSLICES + ks) * 4 + gi) * 2304 + rem];
  const int chr = idxp[sts[gi] + m];
  const int chc = idxp[sts[gi] + n];
  attnS[(size_t)bg * 2304 + rem] = s * invq[b * C_ + chr] * invk[b * C_ + chc];
}

// ================= K6c: row softmax with temperature =================
__global__ void k_softmax(float* __restrict__ attnS, const float* __restrict__ tempr) {
  const int gi = blockIdx.x & 3, b = blockIdx.x >> 2;
  const int gs[4] = {16, 32, 32, 48};
  const int g = gs[gi];
  float* S = attnS + (size_t)(b * 4 + gi) * 2304;
  const float T = tempr[gi];
  const int r = threadIdx.x;
  if (r < g) {
    float mx = -3.4e38f;
    for (int d = 0; d < g; ++d) mx = fmaxf(mx, S[r * 48 + d] * T);
    float sum = 0.0f;
    for (int d = 0; d < g; ++d) {
      float e = expf(S[r * 48 + d] * T - mx);
      S[r * 48 + d] = e; sum += e;
    }
    float inv = 1.0f / sum;
    for (int d = 0; d < g; ++d) S[r * 48 + d] *= inv;
  }
}

// ================= K7: fused attn-apply + sxy + gate/gelu + down + up + proj =================
constexpr size_t SMEM7 = (size_t)(128 * KP + 64 * KP) * 2 + (size_t)(2 * 128 * NP) * 4;

__global__ void k_fused(const __bf16* __restrict__ qkvh, const int* __restrict__ idxp,
                        const float* __restrict__ invq, const float* __restrict__ invk,
                        const float* __restrict__ attnS, const __bf16* __restrict__ wh,
                        const float* __restrict__ b_gate, const float* __restrict__ b_down,
                        const float* __restrict__ b_up, float* __restrict__ out) {
  extern __shared__ __align__(16) char smem[];
  __bf16* aW   = (__bf16*)smem;                               // 128 x KP   (A operand)
  __bf16* bT   = (__bf16*)(smem + 128 * KP * 2);              // 64  x KP   (B operand, N-major)
  float*  outT = (float*)(smem + (128 * KP + 64 * KP) * 2);   // 128 x NP
  float*  sxyT = outT + 128 * NP;                             // 128 x NP
  const unsigned aW_off = (unsigned)(uintptr_t)aW;

  const __bf16* w_gate_h = wh;
  const __bf16* w_down_h = wh + 16384;
  const __bf16* w_up_h   = wh + 24576;
  const __bf16* w_proj_h = wh + 32768;

  const int l0  = blockIdx.x * 64;
  const int b   = blockIdx.y;
  const int tid = threadIdx.x;
  const int w   = tid >> 5;
  int col, rb; frag_coords(col, rb);

  // ---- stage block-diag softmax(attn) into aW (16B zero-fill), V tile into bT ----
  for (int e = tid; e < 128 * KP / 8; e += 256) ((__uint128_t*)aW)[e] = 0;
  for (int seg = tid; seg < 128 * 8; seg += 256) {
    int j = seg >> 3, sgm = seg & 7;
    int ch = idxp[j];
    const v8bf* vrow = (const v8bf*)(qkvh + ((size_t)(b * C3 + 256 + ch)) * L_ + l0);
    v8bf v = vrow[sgm];
#pragma unroll
    for (int i = 0; i < 8; ++i) bT[(sgm * 8 + i) * KP + j] = v[i];
  }
  __syncthreads();
  {
    const int sts[4] = {0, 16, 48, 80};
    const int gsz[4] = {16, 32, 32, 48};
    for (int gi = 0; gi < 4; ++gi) {
      const int g = gsz[gi], st = sts[gi];
      const float* S = attnS + (size_t)(b * 4 + gi) * 2304;
      for (int e = tid; e < g * g; e += 256) {
        int r = e / g, d = e - r * g;
        aW[(st + r) * KP + (st + d)] = (__bf16)S[r * 48 + d];
      }
    }
  }
  __syncthreads();

  // ---- GEMM1: out_all = blockdiag(attn) @ V ----
  v8f acc[4];
  gemm_tiles<8, 4, 4, KP>(aW, bT, acc);
  __syncthreads();
#pragma unroll
  for (int q = 0; q < 4; ++q)
#pragma unroll
    for (int r = 0; r < 8; ++r)
      outT[(w * 16 + rb + r) * NP + q * 16 + col] = acc[q][r];
  __syncthreads();

  // ---- TDM: w_gate -> aW (overlapped with sxy phase) ----
  if (w == 0) tdm_load_2d(w_gate_h, aW_off, 128, 128, 128, 5u, 3u);
  // ---- sxy = out + (qn + kn); 16B loads, write f32 copy + bf16 B operand ----
  {
    int j = tid >> 1, half = tid & 1;
    int ch = idxp[j];
    float iq = invq[b * C_ + ch], ik = invk[b * C_ + ch];
    const v8bf* q8 = (const v8bf*)(qkvh + ((size_t)(b * C3 + ch)) * L_ + l0 + half * 32);
    const v8bf* k8 = (const v8bf*)(qkvh + ((size_t)(b * C3 + 128 + ch)) * L_ + l0 + half * 32);
#pragma unroll
    for (int blk = 0; blk < 4; ++blk) {
      v8bf qv = q8[blk], kv = k8[blk];
#pragma unroll
      for (int i = 0; i < 8; ++i) {
        int n = half * 32 + blk * 8 + i;
        float tc = (float)qv[i] * iq + (float)kv[i] * ik;
        float sv = outT[j * NP + n] + tc;
        sxyT[j * NP + n] = sv;
        bT[n * KP + j]   = (__bf16)sv;
      }
    }
  }
  if (w == 0) __builtin_amdgcn_s_wait_tensorcnt(0);
  __syncthreads();

  // ---- GEMM2: gate = W_g @ sxy; gated = gelu(gate + b_g) * sxy ----
  gemm_tiles<8, 4, 4, KP>(aW, bT, acc);
  __syncthreads();
  if (w == 0) tdm_load_2d(w_down_h, aW_off, 64, 128, 128, 5u, 3u);
#pragma unroll
  for (int q = 0; q < 4; ++q)
#pragma unroll
    for (int r = 0; r < 8; ++r) {
      int m = w * 16 + rb + r, n = q * 16 + col;
      float gt = acc[q][r] + b_gate[m];
      float ge = 0.5f * gt * (1.0f + erff(gt * 0.70710678118654752f));
      bT[n * KP + m] = (__bf16)(ge * sxyT[m * NP + n]);
    }
  if (w == 0) __builtin_amdgcn_s_wait_tensorcnt(0);
  __syncthreads();

  // ---- GEMM3: mod1 = W_down(64x128) @ gated + b_down ----
  gemm_tiles<4, 4, 4, KP>(aW, bT, acc);
  __syncthreads();
  if (w == 0) tdm_load_2d(w_up_h, aW_off, 128, 64, 64, 4u, 35u);
#pragma unroll
  for (int q = 0; q < 2; ++q) {
    int s = w * 2 + q, mt = s >> 2, nt = s & 3;
#pragma unroll
    for (int r = 0; r < 8; ++r) {
      int m = mt * 16 + rb + r, n = nt * 16 + col;
      bT[n * KP + m] = (__bf16)(acc[q][r] + b_down[m]);
    }
  }
  if (w == 0) __builtin_amdgcn_s_wait_tensorcnt(0);
  __syncthreads();

  // ---- GEMM4: out2 = W_up(128x64) @ mod1 + b_up + out_all ----
  gemm_tiles<8, 4, 2, KP>(aW, bT, acc);
  __syncthreads();
  if (w == 0) tdm_load_2d(w_proj_h, aW_off, 128, 128, 128, 5u, 3u);
#pragma unroll
  for (int q = 0; q < 4; ++q)
#pragma unroll
    for (int r = 0; r < 8; ++r) {
      int m = w * 16 + rb + r, n = q * 16 + col;
      bT[n * KP + m] = (__bf16)(acc[q][r] + b_up[m] + outT[m * NP + n]);
    }
  if (w == 0) __builtin_amdgcn_s_wait_tensorcnt(0);
  __syncthreads();

  // ---- GEMM5: out = W_proj @ out2, staged through LDS -> b128 stores ----
  gemm_tiles<8, 4, 4, KP>(aW, bT, acc);
  __syncthreads();                      // outT dead after GEMM4 epilogue -> reuse as f32 stage
  float* fOut = outT;                   // 128 x 64 f32 (fits: 128*NP >= 128*64)
#pragma unroll
  for (int q = 0; q < 4; ++q)
#pragma unroll
    for (int r = 0; r < 8; ++r)
      fOut[(w * 16 + rb + r) * 64 + q * 16 + col] = acc[q][r];
  __syncthreads();
  float* dst = out + ((size_t)b * C_) * L_ + l0;
  for (int seg = tid; seg < 2048; seg += 256) {
    int m = seg >> 4, sg = seg & 15;
    *(__uint128_t*)(dst + (size_t)m * L_ + sg * 4) =
        *(const __uint128_t*)(fOut + m * 64 + sg * 4);
  }
}

// ================= K8: qv_cache broadcast =================
__global__ void k_qvcache(const float* __restrict__ invq, const float* __restrict__ invk,
                          const float* __restrict__ sumq, const float* __restrict__ sumk,
                          const int* __restrict__ idxp, float* __restrict__ outq) {
  const int cout = blockIdx.x, b = blockIdx.y;
  __shared__ float sval;
  if (threadIdx.x == 0) {
    const int sts[4]  = {0, 16, 48, 80};
    const int gsz[4]  = {16, 32, 32, 48};
    const int lens[4] = {128, 128, 128, 96};
    float a = 0.0f;
    for (int i = 0; i < 4; ++i) {
      if (cout < lens[i]) {
        int pos = sts[i] + (cout % gsz[i]);
        int ch  = idxp[pos];
        a += invq[b * C_ + ch] * sumq[b * C_ + ch] + invk[b * C_ + ch] * sumk[b * C_ + ch];
      }
    }
    sval = a * (1.0f / L_) * 0.25f * 0.9f;
  }
  __syncthreads();
  const float v = sval;
  float4 vv = make_float4(v, v, v, v);
  float4* dst = (float4*)(outq + ((size_t)(b * C_ + cout)) * L_);
  for (int e = threadIdx.x; e < L_ / 4; e += 256) dst[e] = vv;
}

// ================= host launch =================
extern "C" void kernel_launch(void* const* d_in, const int* in_sizes, int n_in,
                              void* d_out, int out_size, void* d_ws, size_t ws_size,
                              hipStream_t stream) {
  (void)in_sizes; (void)n_in; (void)out_size; (void)ws_size;
  const float* x      = (const float*)d_in[0];
  const float* tempr  = (const float*)d_in[1];
  const float* w_qkv  = (const float*)d_in[2];
  const float* w_dw   = (const float*)d_in[3];
  const float* w_proj = (const float*)d_in[4];
  const float* w_gate = (const float*)d_in[5];
  const float* b_gate = (const float*)d_in[6];
  const float* w_down = (const float*)d_in[7];
  const float* b_down = (const float*)d_in[8];
  const float* w_up   = (const float*)d_in[9];
  const float* b_up   = (const float*)d_in[10];
  float* out = (float*)d_out;
  float* ws  = (float*)d_ws;

  __bf16* qkv0h = (__bf16*)(ws + W_QKV0H);
  __bf16* qkvh  = (__bf16*)(ws + W_QKVH);
  float*  convP = ws + W_CONVP;
  int*    idxp  = (int*)(ws + W_IDX);
  float*  invq  = ws + W_INVQ;
  float*  invk  = ws + W_INVK;
  float*  sumq  = ws + W_SUMQ;
  float*  sumk  = ws + W_SUMK;
  float*  attnS = ws + W_ATTN;
  float*  attnP = ws + W_ATTNP;
  __bf16* wh    = (__bf16*)(ws + W_WH);

  k_cvtw<<<192, 256, 0, stream>>>(w_gate, w_down, w_up, w_proj, wh);
  k_qkv<<<dim3(L_ / 64, C3 / 64, B_), 256, 0, stream>>>(x, w_qkv, qkv0h);
  k_dwconv<<<(int)(((size_t)B_ * C3 * L_) / 1024), 256, 0, stream>>>(qkv0h, w_dw, qkvh, convP);
  k_rank<<<1, 128, 0, stream>>>(convP, idxp);
  k_norms<<<dim3(C_, 2, B_), 256, 0, stream>>>(qkvh, invq, invk, sumq, sumk);
  k_attn<<<dim3(4, KSLICES, B_), 256, 0, stream>>>(qkvh, idxp, attnP);
  k_attnred<<<(B_ * 4 * 2304 + 255) / 256, 256, 0, stream>>>(attnP, idxp, invq, invk, attnS);
  k_softmax<<<B_ * 4, 64, 0, stream>>>(attnS, tempr);
  k_fused<<<dim3(L_ / 64, B_), 256, SMEM7, stream>>>(qkvh, idxp, invq, invk, attnS,
                                                     wh, b_gate, b_down, b_up, out);
  k_qvcache<<<dim3(C_, B_), 256, 0, stream>>>(invq, invk, sumq, sumk, idxp,
                                              out + (size_t)B_ * C_ * L_);
}
